// SelfAttentionHead_21388937134302
// MI455X (gfx1250) — matrix operand order repaired and checked by reading, not compiled, not run on hardware
//
#include <hip/hip_runtime.h>
#include <hip/hip_bf16.h>

typedef __bf16 bf16_t;
typedef __attribute__((ext_vector_type(8)))  __bf16 v8bf;
typedef __attribute__((ext_vector_type(16))) __bf16 v16bf;
typedef __attribute__((ext_vector_type(8)))  float  v8f;

#define MODEL_C 1024
#define SEQ_T   4096
#define NBATCH  4
#define HEAD_D  64

// ---------------------------------------------------------------------------
// WMMA helpers (CDNA5 bf16 16x16x32, wave32)
// ---------------------------------------------------------------------------
__device__ __forceinline__ v8f bf16_wmma(v16bf a, v16bf b, v8f c) {
  return __builtin_amdgcn_wmma_f32_16x16x32_bf16(
      /*neg_a=*/false, a, /*neg_b=*/false, b,
      /*c_mod=*/(short)0, c, /*reuse_a=*/false, /*reuse_b=*/false);
}

__device__ __forceinline__ v16bf cat16(v8bf lo, v8bf hi) {
  return __builtin_shufflevector(lo, hi, 0, 1, 2, 3, 4, 5, 6, 7,
                                 8, 9, 10, 11, 12, 13, 14, 15);
}

// A-matrix 16x32 (MxK), row-major source with leading dim ld (elements).
// Lane (l = lane%16 -> M row, h = lane/16): halves[0..7] = K h*8+0..7,
// halves[8..15] = K 16+h*8+0..7  (per ISA 16-bit A layout).
__device__ __forceinline__ v16bf load_a_frag(const bf16_t* base, int ld, int lane) {
  int l = lane & 15, h = lane >> 4;
  const bf16_t* p = base + (size_t)l * ld + h * 8;
  v8bf lo = *(const v8bf*)p;
  v8bf hi = *(const v8bf*)(p + 16);
  return cat16(lo, hi);
}

// B-matrix 32x16 (KxN), source is N-major (each n-row has contiguous K),
// leading dim ldn. Lane (n = lane%16, h = lane/16): halves[0..7] = K h*8+0..7,
// halves[8..15] = K 16+h*8+0..7.
__device__ __forceinline__ v16bf load_b_frag(const bf16_t* base, int ldn, int lane) {
  int n = lane & 15, h = lane >> 4;
  const bf16_t* p = base + (size_t)n * ldn + h * 8;
  v8bf lo = *(const v8bf*)p;
  v8bf hi = *(const v8bf*)(p + 16);
  return cat16(lo, hi);
}

// A-fragment built from an f32 row-major source (convert to bf16 on the fly).
__device__ __forceinline__ v16bf load_a_frag_f32(const float* base, int ld, int lane) {
  int l = lane & 15, h = lane >> 4;
  const float* p = base + (size_t)l * ld + h * 8;
  v16bf r;
#pragma unroll
  for (int j = 0; j < 8; ++j) {
    r[j]     = (bf16_t)p[j];
    r[j + 8] = (bf16_t)p[16 + j];
  }
  return r;
}

// ---------------------------------------------------------------------------
// Kernel 1: convert + transpose weights  W[C][64] f32 -> WT[64][C] bf16
// ---------------------------------------------------------------------------
__global__ void prep_weights_kernel(const float* __restrict__ Wq,
                                    const float* __restrict__ Wk,
                                    const float* __restrict__ Wv,
                                    bf16_t* __restrict__ WqT,
                                    bf16_t* __restrict__ WkT,
                                    bf16_t* __restrict__ WvT) {
  int i = blockIdx.x * blockDim.x + threadIdx.x;  // 0 .. C*64-1
  if (i >= MODEL_C * HEAD_D) return;
  int c = i >> 6;
  int d = i & 63;
  size_t t = (size_t)d * MODEL_C + c;
  WqT[t] = (bf16_t)Wq[i];
  WkT[t] = (bf16_t)Wk[i];
  WvT[t] = (bf16_t)Wv[i];
}

// ---------------------------------------------------------------------------
// Kernel 2: projection  out[row][d] = X[row][:] . W[:][d] + b[d]
// One wave computes 16 output rows. WMMA over K=1024 in chunks of 32.
// transpose_out==0: out is bf16 [B*T][64] row-major (q, k)
// transpose_out==1: out is bf16 [B][64][T]          (v transposed)
// ---------------------------------------------------------------------------
__global__ void proj_kernel(const float* __restrict__ X,
                            const bf16_t* __restrict__ WT,
                            const float* __restrict__ bias,
                            bf16_t* __restrict__ out,
                            int transpose_out) {
  int wave = (blockIdx.x * blockDim.x + threadIdx.x) >> 5;  // 0..1023
  int lane = threadIdx.x & 31;
  int n = lane & 15, h = lane >> 4;
  int row0 = wave * 16;
  if (row0 >= NBATCH * SEQ_T) return;

  v8f acc[4] = {};
#pragma unroll 4
  for (int kc = 0; kc < MODEL_C / 32; ++kc) {
    v16bf a = load_a_frag_f32(X + (size_t)row0 * MODEL_C + kc * 32, MODEL_C, lane);
#pragma unroll
    for (int nt = 0; nt < 4; ++nt) {
      v16bf b = load_b_frag(WT + (size_t)(nt * 16) * MODEL_C + kc * 32, MODEL_C, lane);
      acc[nt] = bf16_wmma(a, b, acc[nt]);
    }
  }

#pragma unroll
  for (int nt = 0; nt < 4; ++nt) {
    float bb = bias[nt * 16 + n];
#pragma unroll
    for (int r = 0; r < 8; ++r) {
      float v = acc[nt][r] + bb;
      int row = row0 + r + 8 * h;   // C layout: lane 0-15 M=r, lane 16-31 M=8+r
      int col = nt * 16 + n;
      if (!transpose_out) {
        out[(size_t)row * HEAD_D + col] = (bf16_t)v;
      } else {
        int bidx = row >> 12;        // /4096
        int tt   = row & 4095;
        out[((size_t)bidx * HEAD_D + col) * SEQ_T + tt] = (bf16_t)v;
      }
    }
  }
}

// ---------------------------------------------------------------------------
// Kernel 3: causal flash attention. One wave = 16 query rows.
// Key blocks of 64 (4 S-tiles per softmax round) to amortize the rescale /
// lane-reduction / LDS-relayout overhead; softmax kept in exp2 domain so
// every exponential is a bare v_exp_f32.
// q,k: bf16 [B*T][64] row-major.  vT: bf16 [B][64][T].  out: f32 [B*T][64].
// ---------------------------------------------------------------------------
__global__ void attn_kernel(const bf16_t* __restrict__ qbuf,
                            const bf16_t* __restrict__ kbuf,
                            const bf16_t* __restrict__ vtbuf,
                            float* __restrict__ out) {
  __shared__ __align__(16) bf16_t psh[4][16 * 64];  // per-wave 16x64 P staging

  int wid  = threadIdx.x >> 5;
  int lane = threadIdx.x & 31;
  int n = lane & 15, h = lane >> 4;
  int wave = blockIdx.x * 4 + wid;          // 0..1023
  int b  = wave >> 8;                        // / (T/16)
  int qb = wave & 255;
  int q0 = qb * 16;

  const bf16_t* qp = qbuf + ((size_t)b * SEQ_T + q0) * HEAD_D;
  const bf16_t* kp = kbuf + (size_t)b * SEQ_T * HEAD_D;
  const bf16_t* vp = vtbuf + (size_t)b * HEAD_D * SEQ_T;

  // Resident q fragments (16 rows x 64 K, two K-chunks of 32)
  v16bf aq0 = load_a_frag(qp, HEAD_D, lane);
  v16bf aq1 = load_a_frag(qp + 32, HEAD_D, lane);

  v8f acc[4] = {};
  float m[8], lsum[8];
#pragma unroll
  for (int r = 0; r < 8; ++r) { m[r] = -1.0e38f; lsum[r] = 0.0f; }

  // scale * log2(e): softmax runs in exp2 domain
  const float c2 = 0.125f * 1.44269504088896340736f;
  int nblk = (q0 >> 6) + 1;                  // 64-key blocks up to the diagonal

  for (int j = 0; j < nblk; ++j) {
    int jb = j * 64;

    // ---- S = q . k^T : 16 x 64 scores as four 16x16 C tiles ----
    v8f s[4] = {{0}, {0}, {0}, {0}};
#pragma unroll
    for (int t = 0; t < 4; ++t) {
      v16bf bk0 = load_b_frag(kp + (size_t)(jb + t * 16) * HEAD_D, HEAD_D, lane);
      s[t] = bf16_wmma(aq0, bk0, s[t]);
      v16bf bk1 = load_b_frag(kp + (size_t)(jb + t * 16) * HEAD_D + 32, HEAD_D, lane);
      s[t] = bf16_wmma(aq1, bk1, s[t]);
    }

    bool diag = (jb + 63 > q0);

    // ---- online softmax (rows live in 16-lane half-waves) ----
#pragma unroll
    for (int r = 0; r < 8; ++r) {
      float x[4];
#pragma unroll
      for (int t = 0; t < 4; ++t) x[t] = s[t][r] * c2;
      if (diag) {
        int qg = q0 + r + 8 * h;
#pragma unroll
        for (int t = 0; t < 4; ++t)
          if (jb + t * 16 + n > qg) x[t] = -1.0e38f;
      }
      float mx = fmaxf(fmaxf(x[0], x[1]), fmaxf(x[2], x[3]));
      mx = fmaxf(mx, __shfl_xor(mx, 1, 32));
      mx = fmaxf(mx, __shfl_xor(mx, 2, 32));
      mx = fmaxf(mx, __shfl_xor(mx, 4, 32));
      mx = fmaxf(mx, __shfl_xor(mx, 8, 32));
      float mnew  = fmaxf(m[r], mx);
      float alpha = exp2f(m[r] - mnew);
      m[r] = mnew;
      float p[4];
#pragma unroll
      for (int t = 0; t < 4; ++t) p[t] = exp2f(x[t] - mnew);
      float rs = (p[0] + p[1]) + (p[2] + p[3]);
      rs += __shfl_xor(rs, 1, 32);
      rs += __shfl_xor(rs, 2, 32);
      rs += __shfl_xor(rs, 4, 32);
      rs += __shfl_xor(rs, 8, 32);
      lsum[r] = lsum[r] * alpha + rs;
#pragma unroll
      for (int t = 0; t < 4; ++t) s[t][r] = p[t];
#pragma unroll
      for (int nt = 0; nt < 4; ++nt) acc[nt][r] *= alpha;
    }

    // ---- C-layout -> A-layout for P via per-wave LDS (DS ops in-order) ----
    bf16_t* pw = &psh[wid][0];
#pragma unroll
    for (int r = 0; r < 8; ++r) {
      int row = r + 8 * h;
#pragma unroll
      for (int t = 0; t < 4; ++t)
        pw[row * 64 + t * 16 + n] = (bf16_t)s[t][r];
    }
    v16bf ap0 = load_a_frag(pw, 64, lane);       // keys jb+0..31
    v16bf ap1 = load_a_frag(pw + 32, 64, lane);  // keys jb+32..63

    // ---- acc += P . V  (V^T rows give contiguous B fragments) ----
#pragma unroll
    for (int nt = 0; nt < 4; ++nt) {
      v16bf bv0 = load_b_frag(vp + (size_t)(nt * 16) * SEQ_T + jb, SEQ_T, lane);
      acc[nt] = bf16_wmma(ap0, bv0, acc[nt]);
      v16bf bv1 = load_b_frag(vp + (size_t)(nt * 16) * SEQ_T + jb + 32, SEQ_T, lane);
      acc[nt] = bf16_wmma(ap1, bv1, acc[nt]);
    }
  }

  // ---- epilogue: out = acc / l ----
  size_t orow0 = (size_t)b * SEQ_T + q0;
#pragma unroll
  for (int nt = 0; nt < 4; ++nt) {
#pragma unroll
    for (int r = 0; r < 8; ++r) {
      float v = acc[nt][r] / lsum[r];
      out[(orow0 + r + 8 * h) * HEAD_D + nt * 16 + n] = v;
    }
  }
}

// ---------------------------------------------------------------------------
// Launch
// ---------------------------------------------------------------------------
extern "C" void kernel_launch(void* const* d_in, const int* in_sizes, int n_in,
                              void* d_out, int out_size, void* d_ws, size_t ws_size,
                              hipStream_t stream) {
  const float* Q  = (const float*)d_in[0];
  const float* K  = (const float*)d_in[1];
  const float* V  = (const float*)d_in[2];
  const float* Wq = (const float*)d_in[3];
  const float* bq = (const float*)d_in[4];
  const float* Wk = (const float*)d_in[5];
  const float* bk = (const float*)d_in[6];
  const float* Wv = (const float*)d_in[7];
  const float* bv = (const float*)d_in[8];

  char* ws = (char*)d_ws;
  const size_t WT_BYTES = (size_t)HEAD_D * MODEL_C * sizeof(bf16_t);         // 128 KB
  const size_t QK_BYTES = (size_t)NBATCH * SEQ_T * HEAD_D * sizeof(bf16_t);  // 2 MB
  bf16_t* WqT = (bf16_t*)(ws);
  bf16_t* WkT = (bf16_t*)(ws + WT_BYTES);
  bf16_t* WvT = (bf16_t*)(ws + 2 * WT_BYTES);
  bf16_t* qb  = (bf16_t*)(ws + 3 * WT_BYTES);
  bf16_t* kb  = (bf16_t*)(ws + 3 * WT_BYTES + QK_BYTES);
  bf16_t* vt  = (bf16_t*)(ws + 3 * WT_BYTES + 2 * QK_BYTES);

  // 1) weights -> bf16, transposed
  prep_weights_kernel<<<(MODEL_C * HEAD_D + 255) / 256, 256, 0, stream>>>(
      Wq, Wk, Wv, WqT, WkT, WvT);

  // 2) projections: 1024 waves, 4 waves per 128-thread block
  proj_kernel<<<256, 128, 0, stream>>>(Q, WqT, bq, qb, 0);
  proj_kernel<<<256, 128, 0, stream>>>(K, WkT, bk, kb, 0);
  proj_kernel<<<256, 128, 0, stream>>>(V, WvT, bv, vt, 1);

  // 3) causal flash attention
  attn_kernel<<<256, 128, 0, stream>>>(qb, kb, vt, (float*)d_out);
}
